// PointNetSetUpConv_9354438770915
// MI455X (gfx1250) — matrix operand order repaired and checked by reading, not compiled, not run on hardware
//
#include <hip/hip_runtime.h>
#include <hip/hip_bf16.h>

typedef _Float16 h16;
typedef __attribute__((ext_vector_type(16))) _Float16 v16h;
typedef __attribute__((ext_vector_type(8)))  float    v8f;
typedef __attribute__((ext_vector_type(4)))  unsigned int u32x4;
typedef __attribute__((ext_vector_type(8)))  int          i32x8;
typedef __attribute__((ext_vector_type(4)))  int          i32x4;

#define BB   8
#define NN1  4096
#define NN2  1024
#define KK   16
#define FF1  256
#define FF2  256
#define CIN0 259     // F2 + 3
#define CPAD0 288    // padded to multiple of 32
#define C1   128
#define C2   128
#define C3   256
#define CCAT 512     // C3 + F1
#define COUT 256
#define MROWS (BB*NN1*KK)   // 524288
#define MPTS  (BB*NN1)      // 32768
#define EPSV 1e-5f

// Tensor Data Mover availability (probe-verified builtins; arity differs per toolchain)
#if __has_builtin(__builtin_amdgcn_tensor_load_to_lds) && __has_builtin(__builtin_amdgcn_s_wait_tensorcnt)
#define USE_TDM 1
#if __has_include(<hip/amd_detail/amd_gfx1250_TDM.h>)
#define TDM_ARGS6 1
#else
#define TDM_ARGS6 0
#endif
#else
#define USE_TDM 0
#endif

// ---------------------------------------------------------------------------
// 1) KNN: per query point, 16 nearest of 1024 candidates (pos2 cached in LDS)
// ---------------------------------------------------------------------------
__global__ __launch_bounds__(256) void knn_kernel(const float* __restrict__ pos1,
                                                  const float* __restrict__ pos2,
                                                  int* __restrict__ idx) {
  __shared__ __attribute__((aligned(16))) float p2[NN2 * 3];
  const int b  = blockIdx.x >> 4;
  const int n1 = ((blockIdx.x & 15) << 8) + threadIdx.x;
  for (int e = threadIdx.x; e < NN2 * 3; e += 256) {
    int c = e / NN2, m = e - c * NN2;                  // pos2 is [B,3,N2]
    p2[m * 3 + c] = pos2[(size_t)b * 3 * NN2 + e];
  }
  __syncthreads();
  const float qx = pos1[(size_t)b * 3 * NN1 + 0 * NN1 + n1];
  const float qy = pos1[(size_t)b * 3 * NN1 + 1 * NN1 + n1];
  const float qz = pos1[(size_t)b * 3 * NN1 + 2 * NN1 + n1];
  float bd[KK]; int bi[KK];
#pragma unroll
  for (int j = 0; j < KK; ++j) { bd[j] = 3.0e38f; bi[j] = 0; }
  for (int m = 0; m < NN2; ++m) {
    float dx = p2[m * 3 + 0] - qx;
    float dy = p2[m * 3 + 1] - qy;
    float dz = p2[m * 3 + 2] - qz;
    float d = dx * dx + dy * dy + dz * dz;
    if (d < bd[KK - 1]) {
      bd[KK - 1] = d; bi[KK - 1] = m;
#pragma unroll
      for (int j = KK - 1; j > 0; --j) {
        if (bd[j] < bd[j - 1]) {
          float td = bd[j]; bd[j] = bd[j - 1]; bd[j - 1] = td;
          int   ti = bi[j]; bi[j] = bi[j - 1]; bi[j - 1] = ti;
        }
      }
    }
  }
  int* op = idx + ((size_t)b * NN1 + n1) * KK;
#pragma unroll
  for (int j = 0; j < KK; ++j) op[j] = bi[j];
}

// ---------------------------------------------------------------------------
// 2) Pack weight [N, Kreal] f32 -> WMMA B fragments, f16.
//    Layout: chunk (ks, nt) of 512 halves; lane l, half j within lane:
//    n = nt*16 + (l&15);  k = ks*32 + (l>>4)*16 + j  (contiguous k order, same as A)
// ---------------------------------------------------------------------------
__global__ void pack_w_kernel(const float* __restrict__ w, h16* __restrict__ out,
                              int N, int Kreal, int Kpad) {
  int e = blockIdx.x * 256 + threadIdx.x;
  if (e >= N * Kpad) return;
  int chunk = e >> 9;
  int r = e & 511;
  int lane = r >> 4, j = r & 15;
  int nt = chunk % (N >> 4);
  int ks = chunk / (N >> 4);
  int n = nt * 16 + (lane & 15);
  int k = ks * 32 + (lane >> 4) * 16 + j;
  out[e] = (h16)((k < Kreal) ? w[(size_t)n * Kreal + k] : 0.f);
}

// ---------------------------------------------------------------------------
// 3) Build x0 [MROWS, CPAD0] f16 = concat(gathered feature2, pos_diff, 0-pad)
// ---------------------------------------------------------------------------
__global__ void build_x0_kernel(const float* __restrict__ feature2,
                                const float* __restrict__ pos2,
                                const float* __restrict__ pos1,
                                const int* __restrict__ idx,
                                h16* __restrict__ x0) {
  const size_t e = (size_t)blockIdx.x * 256 + threadIdx.x;
  const size_t total = (size_t)MROWS * CPAD0;
  if (e >= total) return;
  const int c = (int)(e % CPAD0);
  const size_t row = e / CPAD0;
  const int b  = (int)(row / (NN1 * KK));
  const int r  = (int)(row - (size_t)b * NN1 * KK);
  const int n1 = r >> 4;
  float v = 0.f;
  if (c < FF2 + 3) {
    const int m = idx[row];
    if (c < FF2) {
      v = feature2[((size_t)b * FF2 + c) * NN2 + m];
    } else {
      const int cc = c - FF2;
      v = pos2[((size_t)b * 3 + cc) * NN2 + m] - pos1[((size_t)b * 3 + cc) * NN1 + n1];
    }
  }
  x0[e] = (h16)v;
}

// ---------------------------------------------------------------------------
// 4) WMMA GEMM:  C[M,N] = A[M,Kc] * W  (f16 in, f32 accumulate, f16 out)
//    Block = 8 waves in 4(M) x 2(N) grid; block tile 128 x 64; K step 32.
//    Wave: 2x2 fragments (32 rows x 32 cols). A double-buffered in LDS
//    (TDM DMA when available), B fragments loaded straight from packed global.
// ---------------------------------------------------------------------------
#if USE_TDM
__device__ __forceinline__ void tdm_load_tile(const h16* gsrc, unsigned ldsAddr, int rowStrideElems) {
  unsigned long long ga = (unsigned long long)(size_t)gsrc;
  u32x4 g0;
  g0[0] = 1u;                                     // count=1, user mode
  g0[1] = ldsAddr;                                // lds_addr (bytes)
  g0[2] = (unsigned)(ga & 0xFFFFFFFFu);           // global_addr[31:0]
  g0[3] = (unsigned)((ga >> 32) & 0x01FFFFFFu) | (2u << 30);  // addr[56:32] | type=2
  i32x8 g1;
  // data_size=1 (2B) | pad_enable | pad_interval=3 (16 DW) | pad_amount=3 (4 DW)
  g1[0] = (int)((1u << 16) | (1u << 20) | (3u << 22) | (3u << 25));
  unsigned td0 = 32u, td1 = 128u;                 // tensor dims bound the tile
  g1[1] = (int)((td0 & 0xFFFFu) << 16);           // abar_addr=0 | tensor_dim0 lo
  g1[2] = (int)((td0 >> 16) | ((td1 & 0xFFFFu) << 16));
  g1[3] = (int)((td1 >> 16) | (32u << 16));       // tensor_dim1 hi | tile_dim0=32
  g1[4] = (int)(128u);                            // tile_dim1=128, tile_dim2=0
  g1[5] = (int)((unsigned)rowStrideElems);        // tensor_dim0_stride lo32
  g1[6] = 0;
  g1[7] = 0;
  i32x4 z4 = {0, 0, 0, 0};
#if TDM_ARGS6
  i32x8 z8 = {0, 0, 0, 0, 0, 0, 0, 0};
  __builtin_amdgcn_tensor_load_to_lds(g0, g1, z4, z4, z8, 0);
#else
  __builtin_amdgcn_tensor_load_to_lds(g0, g1, z4, z4, 0);
#endif
}
#endif

__global__ __launch_bounds__(256) void gemm_wmma(const h16* __restrict__ A,
                                                 const h16* __restrict__ Bp,
                                                 h16* __restrict__ C,
                                                 int M, int N, int Kc) {
  __shared__ __attribute__((aligned(16))) h16 As[2][128][40];  // 128x32 (+8 pad) x2
  const int tid  = threadIdx.x;
  const int wave = tid >> 5;
  const int lane = tid & 31;
  const int lrow = lane & 15;
  const int hi   = lane >> 4;
  const int wm   = wave & 3;        // 4 waves along M
  const int wn   = wave >> 2;       // 2 waves along N
  const int tileN = blockIdx.x * 64;
  const int tileM = blockIdx.y * 128;
  const int nK   = Kc >> 5;
  const int ntb  = (tileN >> 4) + wn * 2;   // base B chunk (16-col groups)

  const v8f vzero = {0.f, 0.f, 0.f, 0.f, 0.f, 0.f, 0.f, 0.f};
  v8f acc[2][2];
#pragma unroll
  for (int mi = 0; mi < 2; ++mi)
#pragma unroll
    for (int ni = 0; ni < 2; ++ni) acc[mi][ni] = vzero;

  // ---- stage tile 0 ----
#if USE_TDM
  if (wave == 0) {
    tdm_load_tile(A + (size_t)tileM * Kc, (unsigned)(size_t)(void*)&As[0][0][0], Kc);
    __builtin_amdgcn_s_wait_tensorcnt(0);
  }
#else
#pragma unroll
  for (int i = 0; i < 2; ++i) {
    int e = tid + i * 256;
    int row = e >> 2, ch = e & 3;
    const uint4 v = *(const uint4*)(A + (size_t)(tileM + row) * Kc + ch * 8);
    *(uint4*)&As[0][row][ch * 8] = v;
  }
#endif
  __syncthreads();

  for (int ks = 0; ks < nK; ++ks) {
    const int buf = ks & 1;
    const bool pf = (ks + 1) < nK;

#if USE_TDM
    if (wave == 0 && pf) {
      tdm_load_tile(A + (size_t)tileM * Kc + (size_t)(ks + 1) * 32,
                    (unsigned)(size_t)(void*)&As[buf ^ 1][0][0], Kc);
    }
#else
    uint4 r0 = {0, 0, 0, 0}, r1 = {0, 0, 0, 0};
    if (pf) {
      const size_t kn = (size_t)(ks + 1) * 32;
      {
        int e = tid;
        int row = e >> 2, ch = e & 3;
        r0 = *(const uint4*)(A + (size_t)(tileM + row) * Kc + kn + ch * 8);
      }
      {
        int e = tid + 256;
        int row = e >> 2, ch = e & 3;
        r1 = *(const uint4*)(A + (size_t)(tileM + row) * Kc + kn + ch * 8);
      }
      if (ks + 2 < nK)   // prefetch tile after next into caches
        __builtin_prefetch(A + (size_t)(tileM + (tid >> 1)) * Kc + (size_t)(ks + 2) * 32, 0, 1);
    }
#endif

    // ---- compute on As[buf] ----
    union F { v16h v; uint4 q[2]; };
    F af[2];
#pragma unroll
    for (int mi = 0; mi < 2; ++mi) {
      const int m = wm * 32 + mi * 16 + lrow;
      af[mi].q[0] = *(const uint4*)&As[buf][m][hi * 16];
      af[mi].q[1] = *(const uint4*)&As[buf][m][hi * 16 + 8];
    }
#pragma unroll
    for (int ni = 0; ni < 2; ++ni) {
      const v16h bv = *(const v16h*)(Bp + ((size_t)ks * (N >> 4) + ntb + ni) * 512 + lane * 16);
#pragma unroll
      for (int mi = 0; mi < 2; ++mi) {
        acc[mi][ni] = __builtin_amdgcn_wmma_f32_16x16x32_f16(false, af[mi].v, false, bv,
                                                             (short)0, acc[mi][ni], false, false);
      }
    }

#if USE_TDM
    if (wave == 0 && pf) __builtin_amdgcn_s_wait_tensorcnt(0);
#else
    if (pf) {
      {
        int e = tid;
        int row = e >> 2, ch = e & 3;
        *(uint4*)&As[buf ^ 1][row][ch * 8] = r0;
      }
      {
        int e = tid + 256;
        int row = e >> 2, ch = e & 3;
        *(uint4*)&As[buf ^ 1][row][ch * 8] = r1;
      }
    }
#endif
    __syncthreads();
  }

  // C/D f32 layout: VGPR i -> lanes 0-15: M=i, lanes 16-31: M=8+i; N=lane%16
#pragma unroll
  for (int mi = 0; mi < 2; ++mi)
#pragma unroll
    for (int ni = 0; ni < 2; ++ni) {
      const int gN = tileN + wn * 32 + ni * 16 + lrow;
#pragma unroll
      for (int i = 0; i < 8; ++i) {
        const int gM = tileM + wm * 32 + mi * 16 + hi * 8 + i;
        C[(size_t)gM * N + gN] = (h16)acc[mi][ni][i];
      }
    }
}

// ---------------------------------------------------------------------------
// 5) BN stats + normalize
// ---------------------------------------------------------------------------
__global__ void zero_kernel(float* p, int n) {
  int e = blockIdx.x * 256 + threadIdx.x;
  if (e < n) p[e] = 0.f;
}

__global__ __launch_bounds__(256) void stats_kernel(const h16* __restrict__ X,
                                                    float* __restrict__ stats,
                                                    int M, int N, int rpb) {
  const int t = threadIdx.x;
  const int c = t % N;
  const int step = 256 / N;
  int r = blockIdx.x * rpb + t / N;
  int rend = blockIdx.x * rpb + rpb;
  if (rend > M) rend = M;
  float s = 0.f, s2 = 0.f;
  for (; r < rend; r += step) {
    float v = (float)X[(size_t)r * N + c];
    s += v; s2 += v * v;
  }
  atomicAdd(&stats[c], s);
  atomicAdd(&stats[N + c], s2);
}

__global__ void bnrelu_kernel(h16* __restrict__ X, const float* __restrict__ g,
                              const float* __restrict__ bb,
                              const float* __restrict__ stats,
                              size_t total, int N, float invM) {
  size_t e = (size_t)blockIdx.x * 256 + threadIdx.x;
  if (e >= total) return;
  int c = (int)(e % N);
  float mean = stats[c] * invM;
  float var  = stats[N + c] * invM - mean * mean;
  float y = g[c] * ((float)X[e] - mean) * rsqrtf(var + EPSV) + bb[c];
  X[e] = (h16)fmaxf(y, 0.f);
}

// ---------------------------------------------------------------------------
// 6) Max over K + concat feature1 -> pooled [MPTS, 512] f16
// ---------------------------------------------------------------------------
__global__ void pool_concat_kernel(const h16* __restrict__ h3,
                                   const float* __restrict__ feature1,
                                   h16* __restrict__ pooled) {
  int e = blockIdx.x * 256 + threadIdx.x;
  if (e >= MPTS * CCAT) return;
  int row = e / CCAT, c = e - row * CCAT;
  float v;
  if (c < C3) {
    v = -3.0e38f;
    const h16* p = h3 + ((size_t)row * KK) * C3 + c;
#pragma unroll
    for (int k = 0; k < KK; ++k) v = fmaxf(v, (float)p[k * C3]);
  } else {
    int b = row >> 12, n1 = row & (NN1 - 1);
    v = feature1[((size_t)b * FF1 + (c - C3)) * NN1 + n1];
  }
  pooled[e] = (h16)v;
}

// ---------------------------------------------------------------------------
// 7) Final BN + ReLU, f32 transposed output [B, COUT, N1]
// ---------------------------------------------------------------------------
__global__ void final_bn_kernel(const h16* __restrict__ h4,
                                const float* __restrict__ g,
                                const float* __restrict__ bb,
                                const float* __restrict__ stats,
                                float* __restrict__ out) {
  int e = blockIdx.x * 256 + threadIdx.x;
  if (e >= MPTS * COUT) return;
  int row = e / COUT, c = e - row * COUT;
  const float invM = 1.0f / (float)MPTS;
  float mean = stats[c] * invM;
  float var  = stats[COUT + c] * invM - mean * mean;
  float y = g[c] * ((float)h4[e] - mean) * rsqrtf(var + EPSV) + bb[c];
  y = fmaxf(y, 0.f);
  int b = row >> 12, n1 = row & (NN1 - 1);
  out[((size_t)b * COUT + c) * NN1 + n1] = y;
}

// ---------------------------------------------------------------------------
extern "C" void kernel_launch(void* const* d_in, const int* in_sizes, int n_in,
                              void* d_out, int out_size, void* d_ws, size_t ws_size,
                              hipStream_t stream) {
  (void)in_sizes; (void)n_in; (void)out_size; (void)ws_size;
  const float* pos1 = (const float*)d_in[0];
  const float* pos2 = (const float*)d_in[1];
  const float* feature1 = (const float*)d_in[2];
  const float* feature2 = (const float*)d_in[3];
  const float* w1_0 = (const float*)d_in[4];
  const float* g1_0 = (const float*)d_in[5];
  const float* b1_0 = (const float*)d_in[6];
  const float* w1_1 = (const float*)d_in[7];
  const float* g1_1 = (const float*)d_in[8];
  const float* b1_1 = (const float*)d_in[9];
  const float* w1_2 = (const float*)d_in[10];
  const float* g1_2 = (const float*)d_in[11];
  const float* b1_2 = (const float*)d_in[12];
  const float* w2_0 = (const float*)d_in[13];
  const float* g2_0 = (const float*)d_in[14];
  const float* b2_0 = (const float*)d_in[15];

  char* ws = (char*)d_ws;
  size_t off = 0;
  auto alloc = [&](size_t bytes) -> void* {
    off = (off + 255) & ~(size_t)255;
    void* p = ws + off;
    off += bytes;
    return p;
  };
  int* idx     = (int*)alloc((size_t)MPTS * KK * sizeof(int));
  h16* x0      = (h16*)alloc((size_t)MROWS * CPAD0 * sizeof(h16));
  h16* h1      = (h16*)alloc((size_t)MROWS * C1 * sizeof(h16));
  h16* h2b     = (h16*)alloc((size_t)MROWS * C2 * sizeof(h16));
  h16* h3b     = (h16*)alloc((size_t)MROWS * C3 * sizeof(h16));
  h16* pooled  = (h16*)alloc((size_t)MPTS * CCAT * sizeof(h16));
  h16* h4      = (h16*)alloc((size_t)MPTS * COUT * sizeof(h16));
  h16* W0      = (h16*)alloc((size_t)CPAD0 * C1 * sizeof(h16));
  h16* W1      = (h16*)alloc((size_t)C1 * C2 * sizeof(h16));
  h16* W2      = (h16*)alloc((size_t)C2 * C3 * sizeof(h16));
  h16* W3      = (h16*)alloc((size_t)CCAT * COUT * sizeof(h16));
  float* stats = (float*)alloc(2 * 256 * sizeof(float));

  // 1) KNN indices
  knn_kernel<<<dim3(BB * (NN1 / 256)), dim3(256), 0, stream>>>(pos1, pos2, idx);

  // 2) pack weights to f16 fragment layout
  pack_w_kernel<<<dim3((C1 * CPAD0 + 255) / 256), dim3(256), 0, stream>>>(w1_0, W0, C1, CIN0, CPAD0);
  pack_w_kernel<<<dim3((C2 * C1 + 255) / 256), dim3(256), 0, stream>>>(w1_1, W1, C2, C1, C1);
  pack_w_kernel<<<dim3((C3 * C2 + 255) / 256), dim3(256), 0, stream>>>(w1_2, W2, C3, C2, C2);
  pack_w_kernel<<<dim3((COUT * CCAT + 255) / 256), dim3(256), 0, stream>>>(w2_0, W3, COUT, CCAT, CCAT);

  // 3) gather + concat -> x0
  build_x0_kernel<<<dim3((unsigned)(((size_t)MROWS * CPAD0 + 255) / 256)), dim3(256), 0, stream>>>(
      feature2, pos2, pos1, idx, x0);

  const float invMR = 1.0f / (float)MROWS;

  // 4) layer 0: [524288,288] x [288,128]
  gemm_wmma<<<dim3(C1 / 64, MROWS / 128), dim3(256), 0, stream>>>(x0, W0, h1, MROWS, C1, CPAD0);
  zero_kernel<<<dim3(2), dim3(256), 0, stream>>>(stats, 512);
  stats_kernel<<<dim3(MROWS / 2048), dim3(256), 0, stream>>>(h1, stats, MROWS, C1, 2048);
  bnrelu_kernel<<<dim3((unsigned)(((size_t)MROWS * C1 + 255) / 256)), dim3(256), 0, stream>>>(
      h1, g1_0, b1_0, stats, (size_t)MROWS * C1, C1, invMR);

  // 5) layer 1: [524288,128] x [128,128]
  gemm_wmma<<<dim3(C2 / 64, MROWS / 128), dim3(256), 0, stream>>>(h1, W1, h2b, MROWS, C2, C1);
  zero_kernel<<<dim3(2), dim3(256), 0, stream>>>(stats, 512);
  stats_kernel<<<dim3(MROWS / 2048), dim3(256), 0, stream>>>(h2b, stats, MROWS, C2, 2048);
  bnrelu_kernel<<<dim3((unsigned)(((size_t)MROWS * C2 + 255) / 256)), dim3(256), 0, stream>>>(
      h2b, g1_1, b1_1, stats, (size_t)MROWS * C2, C2, invMR);

  // 6) layer 2: [524288,128] x [128,256]
  gemm_wmma<<<dim3(C3 / 64, MROWS / 128), dim3(256), 0, stream>>>(h2b, W2, h3b, MROWS, C3, C2);
  zero_kernel<<<dim3(2), dim3(256), 0, stream>>>(stats, 512);
  stats_kernel<<<dim3(MROWS / 2048), dim3(256), 0, stream>>>(h3b, stats, MROWS, C3, 2048);
  bnrelu_kernel<<<dim3((unsigned)(((size_t)MROWS * C3 + 255) / 256)), dim3(256), 0, stream>>>(
      h3b, g1_2, b1_2, stats, (size_t)MROWS * C3, C3, invMR);

  // 7) max over K + concat feature1
  pool_concat_kernel<<<dim3((MPTS * CCAT + 255) / 256), dim3(256), 0, stream>>>(h3b, feature1, pooled);

  // 8) final layer: [32768,512] x [512,256]
  gemm_wmma<<<dim3(COUT / 64, MPTS / 128), dim3(256), 0, stream>>>(pooled, W3, h4, MPTS, COUT, CCAT);
  zero_kernel<<<dim3(2), dim3(256), 0, stream>>>(stats, 512);
  stats_kernel<<<dim3(MPTS / 2048), dim3(256), 0, stream>>>(h4, stats, MPTS, COUT, 2048);
  final_bn_kernel<<<dim3((MPTS * COUT + 255) / 256), dim3(256), 0, stream>>>(
      h4, g2_0, b2_0, stats, (float*)d_out);
}
